// Q_Attention_90915867722288
// MI455X (gfx1250) — matrix operand order repaired and checked by reading, not compiled, not run on hardware
//
#include <hip/hip_runtime.h>
#include <hip/hip_bf16.h>

// ---------------------------------------------------------------------------
// Types for CDNA5 WMMA
// ---------------------------------------------------------------------------
typedef _Float16 v8h  __attribute__((ext_vector_type(8)));
typedef _Float16 v16h __attribute__((ext_vector_type(16)));
typedef float    v8f  __attribute__((ext_vector_type(8)));

// Problem constants
#define BB   16
#define NN   512
#define CC   768
#define HH   12
#define DD   64
#define MM   (BB * NN)      // 8192
#define K3   (3 * CC)       // 2304

__device__ __forceinline__ v16h cat8(v8h lo, v8h hi) {
  return __builtin_shufflevector(lo, hi, 0,1,2,3,4,5,6,7,8,9,10,11,12,13,14,15);
}

// Lane-xor via ds_swizzle_b32 bit-mask mode: one DS op, no address VALU.
// offset = {xor_mask[4:0] << 10, or_mask=0, and_mask=0x1F}.
// For M < 16 this equals a width-16 xor shuffle (stays within the half).
template <int M>
__device__ __forceinline__ float swz_xor(float v) {
  return __int_as_float(
      __builtin_amdgcn_ds_swizzle(__float_as_int(v), 0x1F | (M << 10)));
}
// Reduce across the 16 lanes of each half (rows of a C-fragment).
__device__ __forceinline__ float red16_max(float v) {
  v = fmaxf(v, swz_xor<1>(v));
  v = fmaxf(v, swz_xor<2>(v));
  v = fmaxf(v, swz_xor<4>(v));
  v = fmaxf(v, swz_xor<8>(v));
  return v;
}
__device__ __forceinline__ float red16_sum(float v) {
  v += swz_xor<1>(v);
  v += swz_xor<2>(v);
  v += swz_xor<4>(v);
  v += swz_xor<8>(v);
  return v;
}
// Full-wave (32 lane) max reduction.
__device__ __forceinline__ float red32_max(float v) {
  v = fmaxf(v, swz_xor<1>(v));
  v = fmaxf(v, swz_xor<2>(v));
  v = fmaxf(v, swz_xor<4>(v));
  v = fmaxf(v, swz_xor<8>(v));
  v = fmaxf(v, swz_xor<16>(v));
  return v;
}

// A fragment (16x32 f16). Contiguous K order (matches B loaders):
//   lane L (0..15):  M=L,     K = {0..15}
//   lane L (16..31): M=L-16,  K = {16..31}
// base points at tile element [0,0], row-major leading dim ld (halves).
__device__ __forceinline__ v16h load_frag_a_rm(const _Float16* base, int ld, int lane) {
  const _Float16* row = base + (size_t)(lane & 15) * ld + ((lane >> 4) << 4);
  v8h lo = *(const v8h*)(row);
  v8h hi = *(const v8h*)(row + 8);
  return cat8(lo, hi);
}

// B fragment (32x16 f16) where memory holds Bt[n][k] (so B[k][n] = Bt[n][k]),
// i.e. weight matrices stored [Nout, K] row-major. Contiguous K order (as in A):
//   lanes 0..15 hold K=0..15, lanes 16..31 hold K=16..31; N = lane&15.
__device__ __forceinline__ v16h load_frag_b_nk(const _Float16* base, int ld, int lane) {
  const _Float16* row = base + (size_t)(lane & 15) * ld + ((lane >> 4) << 4);
  v8h lo = *(const v8h*)(row);
  v8h hi = *(const v8h*)(row + 8);
  return cat8(lo, hi);
}

// B fragment (32x16 f16) where memory holds B[k][n] directly (LDS V tile),
// row-major leading dim ld (halves). Gather: one half per element (ds loads).
__device__ __forceinline__ v16h load_frag_b_kn_lds(const _Float16* base, int ld,
                                                   int lane) {
  const _Float16* col = base + (lane & 15) + (size_t)((lane >> 4) << 4) * ld;
  v16h b;
#pragma unroll
  for (int e = 0; e < 16; ++e) b[e] = col[(size_t)e * ld];
  return b;
}

__device__ __forceinline__ v8f wmma_f16(v16h a, v16h b, v8f c) {
  return __builtin_amdgcn_wmma_f32_16x16x32_f16(false, a, false, b, (short)0, c,
                                                false, false);
}

// ---------------------------------------------------------------------------
// Fake-quant helpers (per-tensor symmetric 8-bit, scale in uint abs-max slot)
// ---------------------------------------------------------------------------
__device__ __forceinline__ float fq_scale(const unsigned* slot) {
  float mx = __uint_as_float(*slot);
  return fmaxf(mx * (1.0f / 127.0f), 1e-8f);
}
__device__ __forceinline__ float fq_val(float v, float s) {
  float q = rintf(v / s);
  q = fminf(fmaxf(q, -128.0f), 127.0f);
  return q * s;
}

// ---------------------------------------------------------------------------
// Small elementwise kernels
// ---------------------------------------------------------------------------
__global__ void absmax_kernel(const float* __restrict__ x, int n,
                              unsigned* __restrict__ slot) {
  float m = 0.0f;
  for (int i = blockIdx.x * blockDim.x + threadIdx.x; i < n;
       i += gridDim.x * blockDim.x)
    m = fmaxf(m, fabsf(x[i]));
  m = red32_max(m);
  if ((threadIdx.x & 31) == 0) atomicMax(slot, __float_as_uint(m));
}

__global__ void quant_w_kernel(const float* __restrict__ w,
                               _Float16* __restrict__ out, int n,
                               const unsigned* __restrict__ slot) {
  float s = fq_scale(slot);
  for (int i = blockIdx.x * blockDim.x + threadIdx.x; i < n;
       i += gridDim.x * blockDim.x)
    out[i] = (_Float16)fq_val(w[i], s);
}

__global__ void cvt_f16_kernel(const float* __restrict__ x,
                               _Float16* __restrict__ out, int n) {
  for (int i = blockIdx.x * blockDim.x + threadIdx.x; i < n;
       i += gridDim.x * blockDim.x)
    out[i] = (_Float16)x[i];
}

__global__ void fq_f16_kernel(_Float16* __restrict__ x, int n,
                              const unsigned* __restrict__ slot) {
  float s = fq_scale(slot);
  for (int i = blockIdx.x * blockDim.x + threadIdx.x; i < n;
       i += gridDim.x * blockDim.x)
    x[i] = (_Float16)fq_val((float)x[i], s);
}

__global__ void finalize_kernel(const float* __restrict__ o2,
                                float* __restrict__ out, int n,
                                const unsigned* __restrict__ slot) {
  float s = fq_scale(slot);
  for (int i = blockIdx.x * blockDim.x + threadIdx.x; i < n;
       i += gridDim.x * blockDim.x)
    out[i] = fq_val(o2[i], s);
  if (blockIdx.x == 0 && threadIdx.x == 0) out[n] = s;  // a_s
}

// ---------------------------------------------------------------------------
// WMMA GEMM:  out[M, Nout] = A[M, K] * W[Nout, K]^T + bias
// One wave computes a 32x64 tile. OUT_F16 -> f16 output (h), else f32 (o2).
// Also accumulates per-tensor abs-max of the (bias-added) result.
// ---------------------------------------------------------------------------
template <bool OUT_F16>
__global__ void gemm_kernel(const _Float16* __restrict__ A,
                            const _Float16* __restrict__ W,
                            const float* __restrict__ bias,
                            _Float16* __restrict__ outh,
                            float* __restrict__ outf,
                            unsigned* __restrict__ absmax_slot,
                            int M, int K, int Nout) {
  int wid  = threadIdx.x >> 5;
  int lane = threadIdx.x & 31;
  int tilesN = Nout >> 6;                 // 64-wide tiles
  int tiles  = (M >> 5) * tilesN;         // 32-tall tiles
  int tile = blockIdx.x * 8 + wid;
  if (tile >= tiles) return;
  int m0 = (tile / tilesN) << 5;
  int n0 = (tile % tilesN) << 6;

  v8f acc[2][4] = {};
  for (int k0 = 0; k0 < K; k0 += 32) {
    // WGP-scope prefetch of next K-step lines (global_prefetch_b8)
    if (k0 + 32 < K) {
      __builtin_prefetch(A + (size_t)(m0 + (lane & 31)) * K + k0 + 32, 0, 3);
      __builtin_prefetch(W + (size_t)(n0 + (lane & 31)) * K + k0 + 32, 0, 3);
      __builtin_prefetch(W + (size_t)(n0 + 32 + (lane & 31)) * K + k0 + 32, 0, 3);
    }
    v16h a0 = load_frag_a_rm(A + (size_t)m0 * K + k0, K, lane);
    v16h a1 = load_frag_a_rm(A + (size_t)(m0 + 16) * K + k0, K, lane);
#pragma unroll
    for (int ni = 0; ni < 4; ++ni) {
      v16h bf = load_frag_b_nk(W + (size_t)(n0 + ni * 16) * K + k0, K, lane);
      acc[0][ni] = wmma_f16(a0, bf, acc[0][ni]);
      acc[1][ni] = wmma_f16(a1, bf, acc[1][ni]);
    }
  }

  float lmax = 0.0f;
#pragma unroll
  for (int mi = 0; mi < 2; ++mi) {
#pragma unroll
    for (int ni = 0; ni < 4; ++ni) {
      int col = n0 + ni * 16 + (lane & 15);
      float bv = bias[col];
#pragma unroll
      for (int r = 0; r < 8; ++r) {
        int row = m0 + mi * 16 + r + ((lane >> 4) << 3);
        float v = acc[mi][ni][r] + bv;
        lmax = fmaxf(lmax, fabsf(v));
        if (OUT_F16)
          outh[(size_t)row * Nout + col] = (_Float16)v;
        else
          outf[(size_t)row * Nout + col] = v;
      }
    }
  }
  lmax = red32_max(lmax);
  if (lane == 0) atomicMax(absmax_slot, __float_as_uint(lmax));
}

// ---------------------------------------------------------------------------
// Flash-style attention.  h = [8192, 2304] f16 (fake-quantized), layout per
// row: [Q(12 heads x 64) | K(...) | V(...)].  One wave handles one
// (batch, head, 16-query tile); streaming softmax over key blocks of 32.
// V tiles are staged global->LDS with CDNA5 async copies (ASYNCcnt),
// overlapped with the S = Q K^T WMMAs.  Output o1 = [8192, 768] f16.
// ---------------------------------------------------------------------------
__global__ void attn_kernel(const _Float16* __restrict__ h,
                            _Float16* __restrict__ o1,
                            unsigned* __restrict__ absmax_slot) {
  __shared__ __align__(16) _Float16 sP[8][16 * 32];   // 8 KB  (P staging)
  __shared__ __align__(16) _Float16 sV[8][32 * 64];   // 32 KB (V tiles)
  int wid  = threadIdx.x >> 5;
  int lane = threadIdx.x & 31;
  int task = blockIdx.x * 8 + wid;        // exactly 192*32 = 6144 tasks
  int bh = task >> 5;                     // (b, head)
  int q0 = (task & 31) << 4;              // query tile start
  int b    = bh / HH;
  int head = bh % HH;

  const _Float16* Qb = h + (size_t)(b * NN) * K3 + head * DD;
  const _Float16* Kb = Qb + CC;
  const _Float16* Vb = Qb + 2 * CC;
  const float scale = 0.125f;             // D^-0.5 = 64^-0.5

  // LDS byte offset of this wave's V slice (generic addr low 32 bits = LDS off)
  const unsigned ldsV = (unsigned)(size_t)(void*)&sV[wid][0];

  v16h qf0 = load_frag_a_rm(Qb + (size_t)q0 * K3 + 0,  K3, lane);
  v16h qf1 = load_frag_a_rm(Qb + (size_t)q0 * K3 + 32, K3, lane);

  v8f acc[4] = {};
  float mrow[8], lrow[8];
#pragma unroll
  for (int r = 0; r < 8; ++r) { mrow[r] = -1.0e30f; lrow[r] = 0.0f; }

  for (int kb = 0; kb < NN; kb += 32) {   // 16 uniform iterations
    // --- Kick off async global->LDS staging of the V tile [32 x 64] f16.
    // 256 16-byte chunks, 8 per lane.  Completion tracked by ASYNCcnt; the
    // previous iteration's ds reads of sV were consumed by its WMMAs (dscnt
    // waited) before we reach this point, so overwrite is safe.
    {
      const _Float16* gV = Vb + (size_t)kb * K3;
#pragma unroll
      for (int c = 0; c < 8; ++c) {
        int id   = c * 32 + lane;
        int row  = id >> 3;
        int col8 = id & 7;
        unsigned long long ga =
            (unsigned long long)(size_t)(gV + (size_t)row * K3 + col8 * 8);
        unsigned la = ldsV + (unsigned)((row * 64 + col8 * 8) * 2);
        asm volatile("global_load_async_to_lds_b128 %0, %1, off"
                     :: "v"(la), "v"(ga) : "memory");
      }
    }

    // --- S = (Q K^T) for 16 queries x 32 keys (two 16x16 C tiles), using
    // regular VMEM loads (LOADcnt) that overlap the async V staging.
    v8f s0 = {}, s1 = {};
    {
      v16h b00 = load_frag_b_nk(Kb + (size_t)kb * K3 + 0,  K3, lane);
      v16h b01 = load_frag_b_nk(Kb + (size_t)kb * K3 + 32, K3, lane);
      s0 = wmma_f16(qf0, b00, s0);
      s0 = wmma_f16(qf1, b01, s0);
      v16h b10 = load_frag_b_nk(Kb + (size_t)(kb + 16) * K3 + 0,  K3, lane);
      v16h b11 = load_frag_b_nk(Kb + (size_t)(kb + 16) * K3 + 32, K3, lane);
      s1 = wmma_f16(qf0, b10, s1);
      s1 = wmma_f16(qf1, b11, s1);
    }

    // --- Online softmax per row; stash P (16x32 f16) into this wave's LDS.
#pragma unroll
    for (int r = 0; r < 8; ++r) {
      float v0 = s0[r] * scale;
      float v1 = s1[r] * scale;
      float rmax = red16_max(fmaxf(v0, v1));
      float mnew = fmaxf(mrow[r], rmax);
      float corr = __expf(mrow[r] - mnew);
      float p0 = __expf(v0 - mnew);
      float p1 = __expf(v1 - mnew);
      float psum = red16_sum(p0 + p1);
      lrow[r] = lrow[r] * corr + psum;
      mrow[r] = mnew;
#pragma unroll
      for (int t = 0; t < 4; ++t) acc[t][r] *= corr;
      int row = r + ((lane >> 4) << 3);
      sP[wid][row * 32 + (lane & 15)]      = (_Float16)p0;
      sP[wid][row * 32 + 16 + (lane & 15)] = (_Float16)p1;
    }
    __syncthreads();  // uniform; orders LDS P writes before A-layout reads

    // --- Wait for the async V tile, then O += P @ V.
    asm volatile("s_wait_asynccnt 0x0" ::: "memory");
    v16h pf = load_frag_a_rm(&sP[wid][0], 32, lane);
#pragma unroll
    for (int t = 0; t < 4; ++t) {
      v16h vf = load_frag_b_kn_lds(&sV[wid][t * 16], 64, lane);
      acc[t] = wmma_f16(pf, vf, acc[t]);
    }
    __syncthreads();
  }

  // Epilogue: normalize, store o1 [B,N,H,D] as f16, track abs-max.
  float inv[8];
#pragma unroll
  for (int r = 0; r < 8; ++r) inv[r] = 1.0f / lrow[r];
  float lmax = 0.0f;
#pragma unroll
  for (int t = 0; t < 4; ++t) {
#pragma unroll
    for (int r = 0; r < 8; ++r) {
      float v = acc[t][r] * inv[r];
      lmax = fmaxf(lmax, fabsf(v));
      int row = q0 + r + ((lane >> 4) << 3);
      int col = head * DD + t * 16 + (lane & 15);
      o1[(size_t)(b * NN + row) * CC + col] = (_Float16)v;
    }
  }
  lmax = red32_max(lmax);
  if (lane == 0) atomicMax(absmax_slot, __float_as_uint(lmax));
}

// ---------------------------------------------------------------------------
// Host side: orchestrate the pipeline on `stream` (graph-capture safe).
// ---------------------------------------------------------------------------
extern "C" void kernel_launch(void* const* d_in, const int* in_sizes, int n_in,
                              void* d_out, int out_size, void* d_ws,
                              size_t ws_size, hipStream_t stream) {
  (void)in_sizes; (void)n_in; (void)out_size; (void)ws_size;

  const float* x      = (const float*)d_in[0];
  const float* qkv_w  = (const float*)d_in[1];
  const float* qkv_b  = (const float*)d_in[2];
  const float* proj_w = (const float*)d_in[3];
  const float* proj_b = (const float*)d_in[4];
  float* out = (float*)d_out;

  // Workspace layout (bytes, 256-aligned):
  //   [0,64)            : 5 uint abs-max slots (w_qkv, w_proj, h, o1, o2)
  //   xh  f16 [8192,768]  : 12,582,912
  //   wq  f16 [2304,768]  :  3,538,944
  //   wp  f16 [768,768]   :  1,179,648
  //   h   f16 [8192,2304] : 37,748,736   (o2 f32 [8192,768] aliases this)
  //   o1  f16 [8192,768]  : 12,582,912
  char* ws = (char*)d_ws;
  unsigned* slots = (unsigned*)ws;
  _Float16* xh = (_Float16*)(ws + 256);
  _Float16* wq = (_Float16*)(ws + 256 + 12582912u);
  _Float16* wp = (_Float16*)(ws + 256 + 12582912u + 3538944u);
  _Float16* hb = (_Float16*)(ws + 256 + 12582912u + 3538944u + 1179648u);
  float*    o2 = (float*)hb;  // alias: h dead after attention
  _Float16* o1 = (_Float16*)(ws + 256 + 12582912u + 3538944u + 1179648u +
                             37748736u);

  const int nW1 = K3 * CC;        // 1,769,472
  const int nW2 = CC * CC;        //   589,824
  const int nX  = MM * CC;        // 6,291,456
  const int nH  = MM * K3;        // 18,874,368

  hipMemsetAsync(slots, 0, 64, stream);

  absmax_kernel<<<512, 256, 0, stream>>>(qkv_w, nW1, slots + 0);
  absmax_kernel<<<256, 256, 0, stream>>>(proj_w, nW2, slots + 1);
  quant_w_kernel<<<(nW1 + 255) / 256, 256, 0, stream>>>(qkv_w, wq, nW1, slots + 0);
  quant_w_kernel<<<(nW2 + 255) / 256, 256, 0, stream>>>(proj_w, wp, nW2, slots + 1);
  cvt_f16_kernel<<<(nX + 255) / 256, 256, 0, stream>>>(x, xh, nX);

  // GEMM1: h = x @ wq^T + qkv_b ; abs-max -> slot2
  gemm_kernel<true><<<(MM / 32) * (K3 / 64) / 8, 256, 0, stream>>>(
      xh, wq, qkv_b, hb, nullptr, slots + 2, MM, CC, K3);
  fq_f16_kernel<<<(nH + 255) / 256, 256, 0, stream>>>(hb, nH, slots + 2);

  // Attention: 192 (b,head) x 32 q-tiles = 6144 waves = 768 blocks
  attn_kernel<<<768, 256, 0, stream>>>(hb, o1, slots + 3);
  fq_f16_kernel<<<(nX + 255) / 256, 256, 0, stream>>>(o1, nX, slots + 3);

  // GEMM2: o2 = o1 @ wp^T + proj_b ; abs-max -> slot4
  gemm_kernel<false><<<(MM / 32) * (CC / 64) / 8, 256, 0, stream>>>(
      o1, wp, proj_b, nullptr, o2, slots + 4, MM, CC, CC);

  finalize_kernel<<<(nX + 255) / 256, 256, 0, stream>>>(o2, out, nX, slots + 4);
}
